// ShowAttendTellModel_43619687858962
// MI455X (gfx1250) — matrix operand order repaired and checked by reading, not compiled, not run on hardware
//
#include <hip/hip_runtime.h>

// ---------------------------------------------------------------------------
// Show-Attend-Tell decoder for MI455X (gfx1250, wave32).
// - All GEMM operands bf16 (converted once), f32 accumulate via
//   v_wmma_f32_16x16x32_bf16 (2x2 frags / wave, 128x64 workgroup tile).
// - Tiles staged with global_load_async_to_lds_b128 + s_wait_asynccnt,
//   double-buffered so WMMA overlaps the async DMA of the next K-slab.
// - ctx_enc/xt_emb kept bf16 (103 MB) so 20 per-step re-reads are L2-resident.
// ---------------------------------------------------------------------------

typedef __attribute__((ext_vector_type(16))) __bf16 v16bf;
typedef __attribute__((ext_vector_type(8)))  float  v8f;
union bfpair { unsigned u; __bf16 h[2]; };

static constexpr int kB  = 256;
static constexpr int kH  = 512;
static constexpr int kE  = 512;
static constexpr int kV  = 10000;
static constexpr int kVp = 10048;          // kV padded to 64
static constexpr int kT  = 20;
static constexpr int kC  = 1024;
static constexpr int kL  = 196;
static constexpr int kM  = kB * kL;        // 50176 (multiple of 128)
static constexpr int kXK = 1536;           // fused gates K = 2E + H
static constexpr int k4H = 2048;

__device__ __forceinline__ float sigmoidf_(float x) { return 1.f / (1.f + __expf(-x)); }

#define TM 128
#define TN 64
#define TK 32
#define LDA 40   // LDS row pitch in bf16: 80B = 16B-aligned chunks, bank-spread

// Async copy of 16 bytes global->LDS (ASYNCcnt-tracked, per ISA 10.x / 15.18.3)
__device__ __forceinline__ void async_b128(unsigned ldsoff, const void* g) {
  asm volatile("global_load_async_to_lds_b128 %0, %1, off"
               :: "v"(ldsoff), "v"((unsigned long long)(size_t)g) : "memory");
}
__device__ __forceinline__ void wait_async0() {
  asm volatile("s_wait_asynccnt 0x0" ::: "memory");
}

// ---------------------------------------------------------------------------
// C[M,N] = epi(A[M,K](bf16) * Bt[Npad,K](bf16)^T + bias)
// M % 128 == 0, Npad % 64 == 0, K % 32 == 0  -> no guards on loads.
// ---------------------------------------------------------------------------
template <bool CHECK_N, bool C_BF16, bool TANH_EPI>
__global__ __launch_bounds__(256) void gemm_async(
    const void* __restrict__ A_, const void* __restrict__ Bt_,
    const float* __restrict__ bias, void* __restrict__ Cout,
    int M, int N, int K)
{
  const __bf16* A  = (const __bf16*)A_;
  const __bf16* Bt = (const __bf16*)Bt_;

  __shared__ alignas(16) __bf16 As[2][TM][LDA];
  __shared__ alignas(16) __bf16 Bs[2][TN][LDA];

  const int tid  = threadIdx.x;
  const int lane = tid & 31;
  const int wave = tid >> 5;
  const int m0 = blockIdx.y * TM;
  const int n0 = blockIdx.x * TN;
  const int wm = (wave >> 1) * 32;   // 4 waves along M
  const int wn = (wave & 1) * 32;    // 2 waves along N

  // A: 128 rows x 4 chunks(16B) = 512 ids (2/thread). B: 64x4 = 256 ids.
  auto issue = [&](int buf, int k0) {
    #pragma unroll
    for (int i = 0; i < 2; ++i) {
      int id = tid + i * 256;
      int r = id >> 2, c = id & 3;
      async_b128((unsigned)(size_t)&As[buf][r][c * 8],
                 A + (size_t)(m0 + r) * K + k0 + c * 8);
    }
    {
      int r = tid >> 2, c = tid & 3;
      async_b128((unsigned)(size_t)&Bs[buf][r][c * 8],
                 Bt + (size_t)(n0 + r) * K + k0 + c * 8);
    }
  };

  v8f acc[2][2] = {};
  const int nk = K / TK;
  issue(0, 0);

  const int rowA   = lane & 15;
  const int kbaseA = (lane >> 4) * 8;    // A 16x32 frag: lanes 16-31 hold K+8 half
  const int colB   = lane & 15;
  const int kbaseB = (lane >> 4) * 8;    // B 32x16 frag: lanes 16-31 hold K+8 half

  for (int ks = 0; ks < nk; ++ks) {
    wait_async0();
    __syncthreads();                       // stage ks visible to all waves;
                                           // stage ks-1 fully consumed by all
    if (ks + 1 < nk) issue((ks + 1) & 1, (ks + 1) * TK);
    const int p = ks & 1;

    v16bf afrag[2], bfrag[2];
    #pragma unroll
    for (int fm = 0; fm < 2; ++fm) {
      int r = wm + fm * 16 + rowA;
      #pragma unroll
      for (int q8 = 0; q8 < 8; ++q8) {
        int kp = ((q8 >> 2) * 16) + kbaseA + ((2 * q8) & 7);   // even -> dword pair
        bfpair q; q.u = *reinterpret_cast<const unsigned*>(&As[p][r][kp]);
        afrag[fm][2 * q8] = q.h[0]; afrag[fm][2 * q8 + 1] = q.h[1];
      }
    }
    #pragma unroll
    for (int fn = 0; fn < 2; ++fn) {
      int ccol = wn + fn * 16 + colB;
      #pragma unroll
      for (int q8 = 0; q8 < 8; ++q8) {
        bfpair q; q.u = *reinterpret_cast<const unsigned*>(&Bs[p][ccol][((q8 >> 2) * 16) + kbaseB + ((2 * q8) & 7)]);
        bfrag[fn][2 * q8] = q.h[0]; bfrag[fn][2 * q8 + 1] = q.h[1];
      }
    }
    #pragma unroll
    for (int fm = 0; fm < 2; ++fm)
      #pragma unroll
      for (int fn = 0; fn < 2; ++fn)
        acc[fm][fn] = __builtin_amdgcn_wmma_f32_16x16x32_bf16(
            false, afrag[fm], false, bfrag[fn], (short)0, acc[fm][fn], false, false);
  }

  // store: C/D layout vgpr r -> M = r (+8 for lanes 16-31), N = lane&15
  const int mlo = (lane < 16) ? 0 : 8;
  const int cn  = lane & 15;
  #pragma unroll
  for (int fm = 0; fm < 2; ++fm)
    #pragma unroll
    for (int fn = 0; fn < 2; ++fn) {
      int gn = n0 + wn + fn * 16 + cn;
      if (CHECK_N && gn >= N) continue;
      float b = bias ? bias[gn] : 0.f;
      #pragma unroll
      for (int r = 0; r < 8; ++r) {
        int gm = m0 + wm + fm * 16 + mlo + r;
        float v = acc[fm][fn][r] + b;
        if (TANH_EPI) v = tanhf(v);
        if (C_BF16) ((__bf16*)Cout)[(size_t)gm * N + gn] = (__bf16)v;
        else        ((float*)Cout)[(size_t)gm * N + gn]  = v;
      }
    }
}

// ---------------------------------------------------------------------------
// features [B,C,L] -> img [B*L, C] bf16 (LDS tile transpose)
// ---------------------------------------------------------------------------
__global__ __launch_bounds__(256) void transpose_feat(const float* __restrict__ feat,
                                                      void* __restrict__ img_) {
  __bf16* img = (__bf16*)img_;
  __shared__ float t[32][33];
  int l0 = blockIdx.x * 32, c0 = blockIdx.y * 32, b = blockIdx.z;
  int tx = threadIdx.x & 31, ty = threadIdx.x >> 5;
  #pragma unroll
  for (int i = 0; i < 4; ++i) {
    int c = c0 + ty + i * 8, l = l0 + tx;
    t[ty + i * 8][tx] = (l < kL) ? feat[((size_t)b * kC + c) * kL + l] : 0.f;
  }
  __syncthreads();
  #pragma unroll
  for (int i = 0; i < 4; ++i) {
    int l = l0 + ty + i * 8, c = c0 + tx;
    if (l < kL) img[((size_t)b * kL + l) * kC + c] = (__bf16)t[tx][ty + i * 8];
  }
}

// feat_mean (bf16 out): wave per (b,c) row, lane-strided over L
__global__ __launch_bounds__(256) void feat_mean_k(const float* __restrict__ feat,
                                                   void* __restrict__ fm_) {
  __bf16* fm = (__bf16*)fm_;
  int b = blockIdx.x, lane = threadIdx.x & 31, wave = threadIdx.x >> 5;
  for (int c = wave; c < kC; c += 8) {
    const float* row = feat + ((size_t)b * kC + c) * kL;
    float s = 0.f;
    for (int l = lane; l < kL; l += 32) s += row[l];
    for (int off = 16; off; off >>= 1) s += __shfl_down(s, off, 32);
    if (lane == 0) fm[b * kC + c] = (__bf16)(s * (1.f / kL));
  }
}

// fp32 W[N,K] -> bf16 [Npad,K], zero-padded rows
__global__ void conv_w_nk(const float* __restrict__ W, void* __restrict__ out_,
                          int N, int K, int Npad) {
  __bf16* out = (__bf16*)out_;
  for (size_t i = (size_t)blockIdx.x * 256 + threadIdx.x; i < (size_t)Npad * K;
       i += (size_t)gridDim.x * 256) {
    int n = (int)(i / K);
    out[i] = (n < N) ? (__bf16)W[i] : (__bf16)0.f;
  }
}

// fp32 W[K,N] -> bf16 [N,K] (transposed copy, one-time)
__global__ void conv_w_kn(const float* __restrict__ W, void* __restrict__ out_,
                          int K, int N) {
  __bf16* out = (__bf16*)out_;
  for (size_t i = (size_t)blockIdx.x * 256 + threadIdx.x; i < (size_t)N * K;
       i += (size_t)gridDim.x * 256) {
    int n = (int)(i / K), k = (int)(i % K);
    out[i] = (__bf16)W[(size_t)k * N + n];
  }
}

// Wcat bf16 [2048][1536] = [W_ih | W_hh];  bcat(f32) = b_ih + b_hh
__global__ void fill_wcat_bf(const float* __restrict__ Wih, const float* __restrict__ Whh,
                             const float* __restrict__ bih, const float* __restrict__ bhh,
                             void* __restrict__ Wcat_, float* __restrict__ bcat) {
  __bf16* Wcat = (__bf16*)Wcat_;
  size_t i = (size_t)blockIdx.x * 256 + threadIdx.x;
  if (i < (size_t)k4H * kXK) {
    int n = (int)(i / kXK), k = (int)(i % kXK);
    float v = (k < 2 * kE) ? Wih[(size_t)n * (2 * kE) + k]
                           : Whh[(size_t)n * kH + (k - 2 * kE)];
    Wcat[i] = (__bf16)v;
  }
  if (i < (size_t)k4H) bcat[i] = bih[i] + bhh[i];
}

// xcat[t][b][512:1024] = emb_table[captions[b,t]]  (bf16)
__global__ void fill_xcat_emb(const int* __restrict__ captions,
                              const float* __restrict__ emb,
                              void* __restrict__ xcat_) {
  __bf16* xcat = (__bf16*)xcat_;
  int t = blockIdx.x >> 8, b = blockIdx.x & 255;
  int tok = captions[b * kT + t];
  __bf16* xrow = xcat + ((size_t)t * kB + b) * kXK + kE;
  for (int e = threadIdx.x; e < kE; e += 256)
    xrow[e] = (__bf16)emb[(size_t)tok * kE + e];
}

// ---------------------------------------------------------------------------
// Per-step attention + xcat row assembly [context | xt | h]
// ---------------------------------------------------------------------------
__global__ __launch_bounds__(256) void attention_k(
    const void* __restrict__ ctx_enc_, const void* __restrict__ xt_emb_,
    const float* __restrict__ hproj, const float* __restrict__ w_att,
    const void* __restrict__ hbuf_, float* __restrict__ ctx_vec,
    void* __restrict__ xcat_, int tstep)
{
  const __bf16* ctx_enc = (const __bf16*)ctx_enc_;
  const __bf16* xt_emb  = (const __bf16*)xt_emb_;
  int b = blockIdx.x;
  __shared__ float hp[kH], wa[kH], sc[224], part[8];
  int tid = threadIdx.x, lane = tid & 31, wave = tid >> 5;

  hp[tid] = hproj[b * kH + tid]; hp[tid + 256] = hproj[b * kH + tid + 256];
  wa[tid] = w_att[tid];          wa[tid + 256] = w_att[tid + 256];
  __syncthreads();

  for (int l = wave; l < kL; l += 8) {
    const __bf16* row = ctx_enc + ((size_t)b * kL + l) * kH;
    float s = 0.f;
    #pragma unroll
    for (int i = 0; i < 16; i += 2) {
      int h = lane * 16 + i;
      bfpair q; q.u = *reinterpret_cast<const unsigned*>(row + h);
      s += tanhf((float)q.h[0] + hp[h])     * wa[h];
      s += tanhf((float)q.h[1] + hp[h + 1]) * wa[h + 1];
    }
    for (int off = 16; off; off >>= 1) s += __shfl_down(s, off, 32);
    if (lane == 0) sc[l] = s;
  }
  __syncthreads();

  float v = (tid < kL) ? sc[tid] : -1e30f;
  float mx = v;
  for (int off = 16; off; off >>= 1) mx = fmaxf(mx, __shfl_down(mx, off, 32));
  if (lane == 0) part[wave] = mx;
  __syncthreads();
  float gmax = part[0];
  #pragma unroll
  for (int i = 1; i < 8; ++i) gmax = fmaxf(gmax, part[i]);
  float e = (tid < kL) ? __expf(v - gmax) : 0.f;
  float se = e;
  for (int off = 16; off; off >>= 1) se += __shfl_down(se, off, 32);
  __syncthreads();
  if (lane == 0) part[wave] = se;
  __syncthreads();
  float tot = 0.f;
  #pragma unroll
  for (int i = 0; i < 8; ++i) tot += part[i];
  if (tid < kL) sc[tid] = e / (tot * (float)kL);   // fold .mean(1)
  __syncthreads();

  float a0 = 0.f, a1 = 0.f;
  const __bf16* xb = xt_emb + (size_t)b * kL * kH;
  for (int l = 0; l < kL; ++l) {
    float al = sc[l];
    bfpair q; q.u = *reinterpret_cast<const unsigned*>(xb + (size_t)l * kH + 2 * tid);
    a0 = fmaf(al, (float)q.h[0], a0);
    a1 = fmaf(al, (float)q.h[1], a1);
  }
  ctx_vec[b * kH + 2 * tid]     = a0;
  ctx_vec[b * kH + 2 * tid + 1] = a1;

  unsigned* xr32 = (unsigned*)((__bf16*)xcat_ + ((size_t)tstep * kB + b) * kXK);
  bfpair cp; cp.h[0] = (__bf16)a0; cp.h[1] = (__bf16)a1;
  xr32[tid] = cp.u;                                            // context [0:512)
  const unsigned* h32 = (const unsigned*)((const __bf16*)hbuf_ + (size_t)b * kH);
  xr32[512 + tid] = h32[tid];                                  // h slice [1024:1536)
}

// LSTM pointwise (i,f,g,o) -> c(f32), h(bf16), svec(bf16)=context+h
__global__ __launch_bounds__(256) void lstm_pointwise(
    const float* __restrict__ gates, void* __restrict__ h_, float* __restrict__ c,
    const float* __restrict__ ctx_vec, void* __restrict__ svec_)
{
  __bf16* h = (__bf16*)h_;
  __bf16* svec = (__bf16*)svec_;
  int b = blockIdx.x;
  const float* g = gates + (size_t)b * k4H;
  for (int j = threadIdx.x; j < kH; j += 256) {
    float ig = sigmoidf_(g[j]);
    float fg = sigmoidf_(g[kH + j]);
    float gg = tanhf(g[2 * kH + j]);
    float og = sigmoidf_(g[3 * kH + j]);
    float cn = fg * c[b * kH + j] + ig * gg;
    float hn = og * tanhf(cn);
    c[b * kH + j] = cn;
    h[b * kH + j] = (__bf16)hn;
    svec[b * kH + j] = (__bf16)(ctx_vec[b * kH + j] + hn);
  }
}

// ---------------------------------------------------------------------------
extern "C" void kernel_launch(void* const* d_in, const int* in_sizes, int n_in,
                              void* d_out, int out_size, void* d_ws, size_t ws_size,
                              hipStream_t stream) {
  (void)in_sizes; (void)n_in; (void)out_size; (void)ws_size;

  const float* features = (const float*)d_in[0];
  const int*   captions = (const int*)d_in[1];
  // d_in[2] = seqlen (device scalar); reference runs T==20 -> compiled in.
  const float* emb     = (const float*)d_in[3];
  const float* W_ih    = (const float*)d_in[4];
  const float* W_hh    = (const float*)d_in[5];
  const float* b_ih    = (const float*)d_in[6];
  const float* b_hh    = (const float*)d_in[7];
  const float* W_cls   = (const float*)d_in[8];
  const float* b_cls   = (const float*)d_in[9];
  const float* W_hatt  = (const float*)d_in[10];
  const float* b_hatt  = (const float*)d_in[11];
  const float* w_att   = (const float*)d_in[12];
  const float* im_attw = (const float*)d_in[13];
  const float* w_ctc   = (const float*)d_in[14];
  const float* W_out   = (const float*)d_in[15];
  const float* b_out   = (const float*)d_in[16];
  const float* W_inith = (const float*)d_in[17];
  const float* b_inith = (const float*)d_in[18];
  const float* W_initm = (const float*)d_in[19];
  const float* b_initm = (const float*)d_in[20];
  float* out = (float*)d_out;

  auto up = [](size_t x) { return (x + 255) & ~size_t(255); };
  char* w = (char*)d_ws; size_t o = 0;
  void* img  = w + o; o += up((size_t)kM * kC * 2);      // dead after feature GEMMs
  void* ctxe = w + o; o += up((size_t)kM * kH * 2);
  void* xte  = w + o; o += up((size_t)kM * kH * 2);
  void* fmean = w + o; o += up((size_t)kB * kC * 2);
  // bf16 weight copies (live across whole run -> dedicated region)
  void* Wimg_t = w + o; o += up((size_t)kH * kC * 2);    // image_att_w^T [512,1024]
  void* Wctc_t = w + o; o += up((size_t)kH * kC * 2);
  void* Whatt_b= w + o; o += up((size_t)kH * kH * 2);
  void* Wout_b = w + o; o += up((size_t)kH * kH * 2);
  void* Winh_b = w + o; o += up((size_t)kH * kC * 2);
  void* Winm_b = w + o; o += up((size_t)kH * kC * 2);
  void* Wcls_b = w + o; o += up((size_t)kVp * kH * 2);   // zero-padded rows
  void* Wcat_b = w + o; o += up((size_t)k4H * kXK * 2);
  float* bcat  = (float*)(w + o); o += up((size_t)k4H * 4);

  char* s = (char*)img; size_t so = 0;                   // alias into img
  void*  xcat  = s + so; so += up((size_t)kT * kB * kXK * 2);
  void*  hbuf  = s + so; so += up((size_t)kB * kH * 2);
  void*  svec  = s + so; so += up((size_t)kB * kH * 2);
  void*  outv  = s + so; so += up((size_t)kB * kH * 2);
  float* cbuf  = (float*)(s + so); so += up((size_t)kB * kH * 4);
  float* hproj = (float*)(s + so); so += up((size_t)kB * kH * 4);
  float* ctxv  = (float*)(s + so); so += up((size_t)kB * kH * 4);
  float* gates = (float*)(s + so); so += up((size_t)kB * k4H * 4);

  auto gemm = [&](const void* A, const void* Bt, const float* bias, void* C,
                  int M, int N, int Npad, int K, int mode /*0 f32,1 bf16,2 bf16+tanh,3 f32+chkN*/) {
    dim3 g(Npad / TN, M / TM);
    switch (mode) {
      case 0: gemm_async<false, false, false><<<g, 256, 0, stream>>>(A, Bt, bias, C, M, N, K); break;
      case 1: gemm_async<false, true,  false><<<g, 256, 0, stream>>>(A, Bt, bias, C, M, N, K); break;
      case 2: gemm_async<false, true,  true ><<<g, 256, 0, stream>>>(A, Bt, bias, C, M, N, K); break;
      default:gemm_async<true,  false, false><<<g, 256, 0, stream>>>(A, Bt, bias, C, M, N, K); break;
    }
  };

  // ---- setup: bf16 weight copies + transposes ----
  conv_w_kn<<<2048, 256, 0, stream>>>(im_attw, Wimg_t, kC, kH);
  conv_w_kn<<<2048, 256, 0, stream>>>(w_ctc,   Wctc_t, kC, kH);
  conv_w_nk<<<1024, 256, 0, stream>>>(W_hatt,  Whatt_b, kH, kH, kH);
  conv_w_nk<<<1024, 256, 0, stream>>>(W_out,   Wout_b,  kH, kH, kH);
  conv_w_nk<<<2048, 256, 0, stream>>>(W_inith, Winh_b,  kH, kC, kH);
  conv_w_nk<<<2048, 256, 0, stream>>>(W_initm, Winm_b,  kH, kC, kH);
  conv_w_nk<<<4096, 256, 0, stream>>>(W_cls,   Wcls_b,  kV, kH, kVp);
  fill_wcat_bf<<<(k4H * kXK + 255) / 256, 256, 0, stream>>>(W_ih, W_hh, b_ih, b_hh, Wcat_b, bcat);

  // ---- precompute (105 of ~200 GFLOP) ----
  transpose_feat<<<dim3((kL + 31) / 32, kC / 32, kB), 256, 0, stream>>>(features, img);
  feat_mean_k<<<kB, 256, 0, stream>>>(features, fmean);
  gemm(img, Wimg_t, nullptr, ctxe, kM, kH, kH, kC, 1);   // ctx_enc bf16
  gemm(img, Wctc_t, nullptr, xte,  kM, kH, kH, kC, 1);   // xt_emb  bf16
  // img dead -> aliased scratch live from here
  fill_xcat_emb<<<kT * kB, 256, 0, stream>>>(captions, emb, xcat);
  gemm(fmean, Winh_b, b_inith, hbuf, kB, kH, kH, kC, 1); // h0 (bf16)
  gemm(fmean, Winm_b, b_initm, cbuf, kB, kH, kH, kC, 0); // c0 (f32)

  // ---- recurrent steps ----
  for (int t = 0; t < kT; ++t) {
    gemm(hbuf, Whatt_b, b_hatt, hproj, kB, kH, kH, kH, 0);
    attention_k<<<kB, 256, 0, stream>>>(ctxe, xte, hproj, w_att, hbuf, ctxv, xcat, t);
    gemm((__bf16*)xcat + (size_t)t * kB * kXK, Wcat_b, bcat, gates, kB, k4H, k4H, kXK, 0);
    lstm_pointwise<<<kB, 256, 0, stream>>>(gates, hbuf, cbuf, ctxv, svec);
    gemm(svec, Wout_b, b_out, outv, kB, kH, kH, kH, 2);  // tanh, bf16
    gemm(outv, Wcls_b, b_cls, out + (size_t)t * kB * kV, kB, kV, kVp, kH, 3);
  }
}
